// NODE_acc_2_pos_small_43628277792929
// MI455X (gfx1250) — compile-verified, matrix operand order checked by reading, not run on hardware
//
#include <hip/hip_runtime.h>
#include <cstdint>
#include <cstddef>

// ---------------- types ----------------
typedef __attribute__((ext_vector_type(16))) _Float16 v16h;
typedef __attribute__((ext_vector_type(8)))  _Float16 v8h;
typedef __attribute__((ext_vector_type(8)))  float    v8f;

// ---------------- problem geometry ----------------
#define NB      16
#define WLEN    4096
#define NPOS    65536          // NB*WLEN
#define CCH     256
#define GROUPS  32
#define KSEG    288            // per-tap channel stride (257 real + pad), 9*32
#define KPAD    864            // 3*KSEG
#define NCHUNK  27             // KPAD/32
#define NTILE   64             // positions per workgroup
#define LROW    264            // LDS row stride (halfs), 16B-aligned rows
#define LROWS   66             // NTILE + 2 halo
#define EPSV    1e-5f

// =====================================================================
// Weight pre-pack: cw [256][257][3][3] f32 -> wb [256][864] f16
// wb[co][kw*288+ci] = cw[co][ci][kh=1][kw], zero beyond ci=256
// =====================================================================
__global__ __launch_bounds__(256) void prep_conv_weights(
    const float* __restrict__ cw, _Float16* __restrict__ wb)
{
    int idx = blockIdx.x * 256 + threadIdx.x;       // CCH*KPAD total
    if (idx >= CCH * KPAD) return;
    int co = idx / KPAD, k = idx - co * KPAD;
    int kw = k / KSEG,  ci = k - kw * KSEG;
    float v = 0.f;
    if (ci < 257) v = cw[((co * 257 + ci) * 3 + 1) * 3 + kw];
    wb[idx] = (_Float16)v;
}

// =====================================================================
// Implicit-GEMM conv (257ch x 3tap -> 256ch) via v_wmma_f32_16x16x32_f16
// act: [256][NPOS] f16 (real channels; t-channel synthesized), out f32 CT
// =====================================================================
__global__ __launch_bounds__(256) void conv257_wmma(
    const _Float16* __restrict__ act,   // [256][NPOS]
    const _Float16* __restrict__ wt,    // [256][KPAD]
    const float*    __restrict__ bias,  // [256]
    float tval,
    float* __restrict__ out)            // [256][NPOS]
{
    __shared__ __align__(16) _Float16 xs[LROWS * LROW + 32];
    const int tid = threadIdx.x;
    const int p0  = blockIdx.x * NTILE;           // tile lies in one image
    const int lo  = (p0 / WLEN) * WLEN;
    const int hi  = lo + WLEN;

    // ---- stage input tile: row r <-> flat pos p0-1+r, slot 0 = t-channel
    for (int idx = tid; idx < LROWS * LROW; idx += 256) {
        int ci = idx / LROWS;                      // 0..263
        int r  = idx - ci * LROWS;                 // 0..65
        int pos = p0 - 1 + r;
        _Float16 v = (_Float16)0.f;
        if (pos >= lo && pos < hi) {
            if (ci == 0)        v = (_Float16)tval;
            else if (ci <= 256) v = act[(size_t)(ci - 1) * NPOS + pos];
        }
        xs[r * LROW + ci] = v;
    }
    if (tid < 32) xs[LROWS * LROW + tid] = (_Float16)0.f;   // tail guard
    __syncthreads();

    // ---- wave tiling: 4 waves along M (64 ch each) x 2 along N (32 pos)
    const int wave = tid >> 5, lane = tid & 31;
    const int g = lane >> 4, ln = lane & 15;
    const int m0w = (wave & 3) * 64;
    const int n0w = (wave >> 2) * 32;

    v8f acc[4][2] = {};

    for (int kc = 0; kc < NCHUNK; ++kc) {
        const int kbase = kc * 32;
        const int kw  = kbase / KSEG;              // single tap per chunk
        const int cib = kbase - kw * KSEG;         // 32-aligned

        if (kc + 1 < NCHUNK)                       // global_prefetch_b8
            __builtin_prefetch(wt + (size_t)(m0w + ln) * KPAD + kbase + 32, 0, 1);

        // A fragments: weights, two contiguous b128 loads per lane
        v16h a[4];
#pragma unroll
        for (int mt = 0; mt < 4; ++mt) {
            const _Float16* wp =
                wt + (size_t)(m0w + mt * 16 + ln) * KPAD + kbase + g * 8;
            v8h alo = *(const v8h*)wp;
            v8h ahi = *(const v8h*)(wp + 16);
            a[mt] = __builtin_shufflevector(alo, ahi,
                     0,1,2,3,4,5,6,7,8,9,10,11,12,13,14,15);
        }
        // B fragments: LDS tile, two contiguous ds_load_b128 per lane
        v16h bf[2];
#pragma unroll
        for (int nt = 0; nt < 2; ++nt) {
            const int r = n0w + nt * 16 + ln + kw;       // input row for tap
            const _Float16* bp = xs + r * LROW + cib + g * 8;
            v8h blo = *(const v8h*)bp;
            v8h bhi = *(const v8h*)(bp + 16);
            bf[nt] = __builtin_shufflevector(blo, bhi,
                      0,1,2,3,4,5,6,7,8,9,10,11,12,13,14,15);
        }
#pragma unroll
        for (int mt = 0; mt < 4; ++mt)
#pragma unroll
            for (int nt = 0; nt < 2; ++nt)
                acc[mt][nt] = __builtin_amdgcn_wmma_f32_16x16x32_f16(
                    false, a[mt], false, bf[nt],
                    (short)0, acc[mt][nt], false, false);
    }

    // ---- store D (+bias): VGPR r -> M = r + 8*g ; N = lane%16
#pragma unroll
    for (int mt = 0; mt < 4; ++mt)
#pragma unroll
        for (int nt = 0; nt < 2; ++nt) {
            const int pos = p0 + n0w + nt * 16 + ln;
#pragma unroll
            for (int r = 0; r < 8; ++r) {
                const int co = m0w + mt * 16 + g * 8 + r;
                out[(size_t)co * NPOS + pos] = acc[mt][nt][r] + bias[co];
            }
        }
}

// =====================================================================
// GroupNorm (32 groups) + activation. MODE: 0 relu->f16, 1 elu->f16,
// 2 plain f32 (in-place safe). Block = one (batch, group) slice.
// =====================================================================
template<int MODE>
__global__ __launch_bounds__(256) void groupnorm_k(
    const float* __restrict__ in, const float* __restrict__ gamma,
    const float* __restrict__ beta, float* __restrict__ outF,
    _Float16* __restrict__ outH)
{
    const int bg = blockIdx.x, b = bg >> 5, grp = bg & 31;
    const int tid = threadIdx.x;
    __shared__ float s1[256], s2[256];
    const size_t bpos = (size_t)b * WLEN;

    float sum = 0.f, sq = 0.f;
    for (int c8 = 0; c8 < 8; ++c8) {
        const float* row = in + (size_t)(grp * 8 + c8) * NPOS + bpos;
        for (int w = tid; w < WLEN; w += 256) {
            float v = row[w]; sum += v; sq += v * v;
        }
    }
    s1[tid] = sum; s2[tid] = sq; __syncthreads();
    for (int s = 128; s > 0; s >>= 1) {
        if (tid < s) { s1[tid] += s1[tid + s]; s2[tid] += s2[tid + s]; }
        __syncthreads();
    }
    const float mean = s1[0] * (1.f / 32768.f);
    const float var  = s2[0] * (1.f / 32768.f) - mean * mean;
    const float rstd = rsqrtf(fmaxf(var, 0.f) + EPSV);

    for (int c8 = 0; c8 < 8; ++c8) {
        const int c = grp * 8 + c8;
        const float ga = gamma[c], be = beta[c];
        const float* row = in + (size_t)c * NPOS + bpos;
        for (int w = tid; w < WLEN; w += 256) {
            float v = (row[w] - mean) * rstd * ga + be;
            if (MODE == 0) {
                v = fmaxf(v, 0.f);
                outH[(size_t)c * NPOS + bpos + w] = (_Float16)v;
            } else if (MODE == 1) {
                v = v > 0.f ? v : (expf(v) - 1.f);
                outH[(size_t)c * NPOS + bpos + w] = (_Float16)v;
            } else {
                outF[(size_t)c * NPOS + bpos + w] = v;
            }
        }
    }
}

// =====================================================================
// Stem: conv1d(4->64,k3,pad1), BN+relu, conv1d(64->256,k3), BN+tanh
// =====================================================================
__global__ __launch_bounds__(256) void stem_conv1(
    const float* __restrict__ x, const float* __restrict__ w,
    const float* __restrict__ bias, float* __restrict__ out)
{
    int idx = blockIdx.x * 256 + threadIdx.x;       // 64*NPOS
    int co = idx >> 16, pos = idx & (NPOS - 1);
    int b = pos >> 12, wp = pos & (WLEN - 1);
    float s = bias[co];
#pragma unroll
    for (int ci = 0; ci < 4; ++ci)
#pragma unroll
        for (int kw = 0; kw < 3; ++kw) {
            int wi = wp + kw - 1;
            if (wi >= 0 && wi < WLEN)
                s += x[(b * 4 + ci) * WLEN + wi] * w[(co * 4 + ci) * 3 + kw];
        }
    out[(size_t)co * NPOS + pos] = s;
}

__global__ __launch_bounds__(256) void stem_conv2(
    const float* __restrict__ t1, const float* __restrict__ w,
    const float* __restrict__ bias, float* __restrict__ out)
{
    int idx = blockIdx.x * 256 + threadIdx.x;       // 256*NPOS
    int co = idx >> 16, pos = idx & (NPOS - 1);
    int b = pos >> 12, wp = pos & (WLEN - 1);
    float s = bias[co];
    for (int ci = 0; ci < 64; ++ci) {
        const float* row = t1 + (size_t)ci * NPOS + ((size_t)b << 12);
        const float* wr  = w + (co * 64 + ci) * 3;
        if (wp > 0)        s += row[wp - 1] * wr[0];
        s += row[wp] * wr[1];
        if (wp < WLEN - 1) s += row[wp + 1] * wr[2];
    }
    out[(size_t)co * NPOS + pos] = s;
}

// ACT: 0 relu, 1 tanh ; one block per channel, in-place
template<int ACT>
__global__ __launch_bounds__(256) void stem_bn(
    float* __restrict__ data, const float* __restrict__ g,
    const float* __restrict__ be)
{
    const int c = blockIdx.x, tid = threadIdx.x;
    __shared__ float s1[256], s2[256];
    float* row = data + (size_t)c * NPOS;
    float sum = 0.f, sq = 0.f;
    for (int i = tid; i < NPOS; i += 256) { float v = row[i]; sum += v; sq += v * v; }
    s1[tid] = sum; s2[tid] = sq; __syncthreads();
    for (int s = 128; s > 0; s >>= 1) {
        if (tid < s) { s1[tid] += s1[tid + s]; s2[tid] += s2[tid + s]; }
        __syncthreads();
    }
    const float mean = s1[0] * (1.f / NPOS);
    const float var  = s2[0] * (1.f / NPOS) - mean * mean;
    const float rstd = rsqrtf(fmaxf(var, 0.f) + EPSV);
    const float ga = g[c], bb = be[c];
    for (int i = tid; i < NPOS; i += 256) {
        float v = (row[i] - mean) * rstd * ga + bb;
        row[i] = (ACT == 0) ? fmaxf(v, 0.f) : tanhf(v);
    }
}

// =====================================================================
// Elementwise ODE updates + final layout transpose
// =====================================================================
__global__ __launch_bounds__(256) void axpy_k(
    const float* __restrict__ y, const float* __restrict__ f,
    float h, float* __restrict__ o)
{
    int i = blockIdx.x * 256 + threadIdx.x;
    o[i] = y[i] + h * f[i];
}

__global__ __launch_bounds__(256) void comb_k(
    const float* __restrict__ y, const float* __restrict__ f0,
    const float* __restrict__ ft, float hh, float* __restrict__ o)
{
    int i = blockIdx.x * 256 + threadIdx.x;
    o[i] = y[i] + hh * (f0[i] + ft[i]);
}

__global__ __launch_bounds__(256) void to_out_k(
    const float* __restrict__ yf, float* __restrict__ out)
{
    int i = blockIdx.x * 256 + threadIdx.x;     // CT index c*NPOS+pos
    int c = i >> 16, pos = i & (NPOS - 1);
    int b = pos >> 12, w = pos & (WLEN - 1);
    out[(((size_t)b * CCH + c) << 12) + w] = yf[i];
}

// =====================================================================
// Host orchestration
// =====================================================================
extern "C" void kernel_launch(void* const* d_in, const int* in_sizes, int n_in,
                              void* d_out, int out_size, void* d_ws, size_t ws_size,
                              hipStream_t stream)
{
    (void)in_sizes; (void)n_in; (void)out_size; (void)ws_size;
    const float* x    = (const float*)d_in[0];
    const float* w1   = (const float*)d_in[1];
    const float* b1   = (const float*)d_in[2];
    const float* bn1g = (const float*)d_in[3];
    const float* bn1b = (const float*)d_in[4];
    const float* w2   = (const float*)d_in[5];
    const float* b2   = (const float*)d_in[6];
    const float* bn2g = (const float*)d_in[7];
    const float* bn2b = (const float*)d_in[8];
    const float* cw1  = (const float*)d_in[9];
    const float* cb1  = (const float*)d_in[10];
    const float* gn1g = (const float*)d_in[11];
    const float* gn1b = (const float*)d_in[12];
    const float* gn2g = (const float*)d_in[13];
    const float* gn2b = (const float*)d_in[14];
    const float* cw2  = (const float*)d_in[15];
    const float* cb2  = (const float*)d_in[16];
    const float* gn3g = (const float*)d_in[17];
    const float* gn3b = (const float*)d_in[18];

    // workspace carve-up
    char* ws = (char*)d_ws;
    _Float16* Wb1 = (_Float16*)ws;                       // 256*864*2
    _Float16* Wb2 = Wb1 + (size_t)CCH * KPAD;
    _Float16* actH = (_Float16*)(ws + (1u << 20));       // 32 MB
    float* S0 = (float*)(ws + (1u << 20) + ((size_t)CCH * NPOS * 2));
    float* S1 = S0 + (size_t)CCH * NPOS;                 // 64 MB each
    float* F0 = S1 + (size_t)CCH * NPOS;
    float* FT = F0 + (size_t)CCH * NPOS;
    float* T1 = FT;                                      // stem tmp aliases FT

    const int NE = CCH * NPOS;                           // 16.7M elems

    // pack conv weights to f16 implicit-GEMM layout
    prep_conv_weights<<<(CCH * KPAD + 255) / 256, 256, 0, stream>>>(cw1, Wb1);
    prep_conv_weights<<<(CCH * KPAD + 255) / 256, 256, 0, stream>>>(cw2, Wb2);

    // ---- stem -> y0 in S0 (channel-major CT layout) ----
    stem_conv1<<<64 * NPOS / 256, 256, 0, stream>>>(x, w1, b1, T1);
    stem_bn<0><<<64, 256, 0, stream>>>(T1, bn1g, bn1b);
    stem_conv2<<<NE / 256, 256, 0, stream>>>(T1, w2, b2, S0);
    stem_bn<1><<<CCH, 256, 0, stream>>>(S0, bn2g, bn2b);

    // ---- ODE RHS: f(t, yin) -> fout ----
    auto f_eval = [&](float tval, const float* yin, float* fout) {
        groupnorm_k<0><<<NB * GROUPS, 256, 0, stream>>>(yin, gn1g, gn1b, nullptr, actH);
        conv257_wmma<<<NPOS / NTILE, 256, 0, stream>>>(actH, Wb1, cb1, tval, fout);
        groupnorm_k<1><<<NB * GROUPS, 256, 0, stream>>>(fout, gn2g, gn2b, nullptr, actH);
        conv257_wmma<<<NPOS / NTILE, 256, 0, stream>>>(actH, Wb2, cb2, tval, fout);
        groupnorm_k<2><<<NB * GROUPS, 256, 0, stream>>>(fout, gn3g, gn3b, fout, nullptr);
    };

    // ---- Heun integrator: 8 steps, 2 corrector passes ----
    const float h = 1.f / 8.f;
    float* yc = S0;
    float* yn = S1;
    for (int i = 0; i < 8; ++i) {
        const float t = (float)i * h;
        f_eval(t, yc, F0);
        axpy_k<<<NE / 256, 256, 0, stream>>>(yc, F0, h, yn);
        for (int c = 0; c < 2; ++c) {
            f_eval(t + h, yn, FT);
            comb_k<<<NE / 256, 256, 0, stream>>>(yc, F0, FT, 0.5f * h, yn);
        }
        float* tmp = yc; yc = yn; yn = tmp;
    }

    // ---- CT -> NCHW output ----
    to_out_k<<<NE / 256, 256, 0, stream>>>(yc, (float*)d_out);
}
